// CoordPE_MP_25958782337709
// MI455X (gfx1250) — compile-verified, matrix operand order checked by reading, not run on hardware
//
#include <hip/hip_runtime.h>

// B=2, L=2048, D=128, K=3, NRBF=16
#define LSEQ 2048
#define NBATCH 2
#define DD 128
#define KLAYERS 3
#define NRBF 16

// big-GEMM tiling
#define MROWS 64                  // rows per workgroup M-block
#define KCHUNK 128                // K elements staged per TDM chunk
#define NCHUNKS (LSEQ / KCHUNK)   // 16

typedef _Float16 v16h __attribute__((ext_vector_type(16)));
typedef _Float16 v8h  __attribute__((ext_vector_type(8)));
typedef float    v8f  __attribute__((ext_vector_type(8)));
typedef unsigned int u32x4 __attribute__((ext_vector_type(4)));
typedef int          i32x8 __attribute__((ext_vector_type(8)));
typedef int          i32x4 __attribute__((ext_vector_type(4)));

// -------------------------------------------------------------------------
// Kernel 1: pairwise distances -> W = exp(-d) stored f16, plus RBF row means
// grid (L, B), block 256. One block per output row i.
// -------------------------------------------------------------------------
__global__ void dist_rbf_kernel(const float* __restrict__ coords,
                                const float* __restrict__ gamma_p,
                                const float* __restrict__ centers,
                                _Float16* __restrict__ Wh,
                                float* __restrict__ rbfloc) {
  const int i = blockIdx.x;
  const int b = blockIdx.y;
  const int tid = threadIdx.x;
  __shared__ float red[256 * NRBF];

  const float gamma = gamma_p[0];
  const float cx = coords[((size_t)b * LSEQ + i) * 3 + 0];
  const float cy = coords[((size_t)b * LSEQ + i) * 3 + 1];
  const float cz = coords[((size_t)b * LSEQ + i) * 3 + 2];

  float cen[NRBF];
#pragma unroll
  for (int r = 0; r < NRBF; ++r) cen[r] = centers[r];
  float acc[NRBF];
#pragma unroll
  for (int r = 0; r < NRBF; ++r) acc[r] = 0.0f;

  for (int j = tid; j < LSEQ; j += 256) {
    const float dx = cx - coords[((size_t)b * LSEQ + j) * 3 + 0];
    const float dy = cy - coords[((size_t)b * LSEQ + j) * 3 + 1];
    const float dz = cz - coords[((size_t)b * LSEQ + j) * 3 + 2];
    const float sq = dx * dx + dy * dy + dz * dz;
    const float d = (sq > 0.0f) ? sqrtf(sq) : 0.0f;
    Wh[(((size_t)b * LSEQ + i) * LSEQ) + j] = (_Float16)expf(-d);
#pragma unroll
    for (int r = 0; r < NRBF; ++r) {
      const float t = d - cen[r];
      acc[r] += expf(-gamma * t * t);
    }
  }
#pragma unroll
  for (int r = 0; r < NRBF; ++r) red[tid * NRBF + r] = acc[r];
  __syncthreads();
  for (int s = 128; s > 0; s >>= 1) {
    if (tid < s) {
#pragma unroll
      for (int r = 0; r < NRBF; ++r)
        red[tid * NRBF + r] += red[(tid + s) * NRBF + r];
    }
    __syncthreads();
  }
  if (tid < NRBF)
    rbfloc[((size_t)b * LSEQ + i) * NRBF + tid] = red[tid] * (1.0f / LSEQ);
}

// -------------------------------------------------------------------------
// Kernel 2: encoder  h = [atom_emb[Z] | rbf_local@rbf_w + rbf_b] @ out_w + out_b
// grid (L, B), block 128. Output h in f16 row-major [B*L, 128].
// -------------------------------------------------------------------------
__global__ void encoder_kernel(const int* __restrict__ Z,
                               const float* __restrict__ atom_emb,
                               const float* __restrict__ rbf_w,
                               const float* __restrict__ rbf_b,
                               const float* __restrict__ out_w,
                               const float* __restrict__ out_b,
                               const float* __restrict__ rbfloc,
                               _Float16* __restrict__ h16) {
  const int i = blockIdx.x;
  const int b = blockIdx.y;
  const int t = threadIdx.x;
  const int row = b * LSEQ + i;
  __shared__ float ha[DD];
  __shared__ float hg[DD];

  const int z = Z[row];
  ha[t] = atom_emb[(size_t)z * DD + t];
  float g = rbf_b[t];
#pragma unroll
  for (int r = 0; r < NRBF; ++r)
    g += rbfloc[(size_t)row * NRBF + r] * rbf_w[r * DD + t];
  hg[t] = g;
  __syncthreads();

  float o = out_b[t];
  for (int c = 0; c < DD; ++c) {
    o += ha[c] * out_w[c * DD + t];
    o += hg[c] * out_w[(DD + c) * DD + t];
  }
  h16[(size_t)row * DD + t] = (_Float16)o;
}

// -------------------------------------------------------------------------
// Kernel 3: per-layer weight folding + pack into WMMA B-fragment order.
//   A_k  = msg_w[k]  @ U2 ; Bw_k = self_w[k] @ U1
//   cvec = msg_b[k] @ U2  ; dvec = self_b[k] @ U1 + upd_b[k]
// Fragment layout (ISA 7.12.2, 16-bit B 32x16):
//   lane: N = lane%16, khalf = (lane<16)?0:8 ; element e: K=(e/8)*16+khalf+(e%8)
// -------------------------------------------------------------------------
__global__ void fold_pack_kernel(int k,
                                 const float* __restrict__ self_w,
                                 const float* __restrict__ self_b,
                                 const float* __restrict__ msg_w,
                                 const float* __restrict__ msg_b,
                                 const float* __restrict__ upd_w,
                                 const float* __restrict__ upd_b,
                                 _Float16* __restrict__ Apack,
                                 _Float16* __restrict__ Bpack,
                                 float* __restrict__ cvec,
                                 float* __restrict__ dvec) {
  const int frag = blockIdx.x;          // 0..31 = kt*8+nt
  const int sel = blockIdx.y;           // 0: msg path, 1: self path
  const int kt = frag >> 3, nt = frag & 7;
  const float* Wsrc = sel ? self_w : msg_w;
  const float* U = upd_w + (size_t)k * 2 * DD * DD + (sel ? 0 : DD * DD);
  _Float16* dst = sel ? Bpack : Apack;

  for (int id = threadIdx.x; id < 512; id += blockDim.x) {
    const int lane = id >> 4, e = id & 15;
    const int kk = kt * 32 + ((e >> 3) << 4) + ((lane >= 16) ? 8 : 0) + (e & 7);
    const int n = nt * 16 + (lane & 15);
    const float* wr = Wsrc + ((size_t)k * DD + kk) * DD;
    float s = 0.0f;
    for (int t = 0; t < DD; ++t) s += wr[t] * U[t * DD + n];
    dst[frag * 512 + id] = (_Float16)s;
  }
  if (frag == 0 && threadIdx.x < DD) {
    const int n = threadIdx.x;
    const float* bsrc = (sel ? self_b : msg_b) + (size_t)k * DD;
    float s = 0.0f;
    for (int t = 0; t < DD; ++t) s += bsrc[t] * U[t * DD + n];
    if (sel) s += upd_b[(size_t)k * DD + n];
    (sel ? dvec : cvec)[n] = s;
  }
}

// -------------------------------------------------------------------------
// Kernel 4: medium GEMM  [4096,128] x [128,128] via v_wmma_f32_16x16x32_f16.
// y==0: m = h@A_k + cvec  -> scattered into big-GEMM B-fragment order
// y==1: hb = h@Bw_k + dvec -> f32 row-major
// grid (256, 2), block 256 (8 waves; wave w = n-tile w).
// -------------------------------------------------------------------------
__global__ void med_gemm_kernel(const _Float16* __restrict__ h16,
                                const _Float16* __restrict__ Apack,
                                const _Float16* __restrict__ Bpack,
                                const float* __restrict__ cvec,
                                const float* __restrict__ dvec,
                                _Float16* __restrict__ mpack,
                                float* __restrict__ hb32) {
  const int mtile = blockIdx.x;
  const int sel = blockIdx.y;
  const int wave = threadIdx.x >> 5;  // n-tile
  const int lane = threadIdx.x & 31;
  const int nIn = lane & 15;
  const int khalf = (lane < 16) ? 0 : 8;
  const _Float16* Bp = sel ? Bpack : Apack;
  const _Float16* arow = h16 + (size_t)(mtile * 16 + nIn) * DD + khalf;

  v8f acc = {};
#pragma unroll
  for (int kt = 0; kt < 4; ++kt) {
    v8h lo = *(const v8h*)(arow + kt * 32);
    v8h hi = *(const v8h*)(arow + kt * 32 + 16);
    v16h a = __builtin_shufflevector(lo, hi, 0, 1, 2, 3, 4, 5, 6, 7, 8, 9, 10,
                                     11, 12, 13, 14, 15);
    v16h bf = *(const v16h*)(Bp + ((kt * 8 + wave) * 32 + lane) * 16);
    acc = __builtin_amdgcn_wmma_f32_16x16x32_f16(false, a, false, bf, (short)0,
                                                 acc, false, false);
  }
  const float bias = (sel ? dvec : cvec)[wave * 16 + nIn];
  if (sel) {
#pragma unroll
    for (int v = 0; v < 8; ++v) {
      const int row = mtile * 16 + v + 8 * (lane >> 4);
      hb32[(size_t)row * DD + wave * 16 + nIn] = acc[v] + bias;
    }
  } else {
#pragma unroll
    for (int v = 0; v < 8; ++v) {
      const int row = mtile * 16 + v + 8 * (lane >> 4);
      const int b = row >> 11, ii = row & (LSEQ - 1);
      const int kt2 = ii >> 5, krem = ii & 31;
      const int g = (krem >> 4) & 1, hi8 = (krem >> 3) & 1, rr = krem & 7;
      const int lp = hi8 * 16 + nIn;
      const int e = g * 8 + rr;
      mpack[(size_t)(((b * 64 + kt2) * 8 + wave) * 32 + lp) * 16 + e] =
          (_Float16)(acc[v] + bias);
    }
  }
}

// -------------------------------------------------------------------------
// TDM: stage a MROWS x KCHUNK f16 tile of the (LSEQ x LSEQ) W tensor into LDS.
// D# per ISA ch8: group0 = {count=1 | lds_addr | 57b global tile addr | type=2}
//                 group1 = {data_size=2B, tdim0/1=LSEQ, tile0=KCHUNK,
//                           tile1=MROWS, dim0_stride=LSEQ}; groups2/3 = 0 (2D).
// clang-23 lane: 6-arg builtin (u32x4, i32x8, i32x4, i32x4, i32x8, i32 cpol).
// Completion tracked with TENSORcnt (s_wait_tensorcnt).
// -------------------------------------------------------------------------
__device__ __forceinline__ void tdm_load_tile(const _Float16* gsrc,
                                              unsigned lds_byte_off) {
  const unsigned long long ga = (unsigned long long)(size_t)gsrc;
  u32x4 g0 = {1u,                                   // count=1, no gather
              lds_byte_off,                         // lds_addr
              (unsigned)(ga & 0xFFFFFFFFu),
              (unsigned)((ga >> 32) & 0x1FFFFFFu) | (2u << 30)};  // type=2
  i32x8 g1 = {(1 << 16),                            // data_size = 2 bytes
              (int)((LSEQ & 0xFFFF) << 16),         // tensor_dim0 lo16
              (int)((LSEQ & 0xFFFF) << 16),         // dim0 hi16(0) | dim1 lo16
              (int)(KCHUNK << 16),                  // dim1 hi16(0) | tile_dim0
              MROWS,                                // tile_dim1 | tile_dim2=0
              LSEQ,                                 // tensor_dim0_stride lo32
              0, 0};
  i32x4 gz4 = {0, 0, 0, 0};
  i32x8 gz8 = {0, 0, 0, 0, 0, 0, 0, 0};
  __builtin_amdgcn_tensor_load_to_lds(g0, g1, gz4, gz4, gz8, 0);
}

// -------------------------------------------------------------------------
// Kernel 5: big GEMM  msg = W[b] @ m[b]  (2048x2048 x 2048x128), K=2048.
// One WG = 64-row M-block, 8 waves (one n-tile each), 4 accumulators/wave.
// W tiles DMA'd into LDS by the Tensor Data Mover, double-buffered.
// Epilogue: h' = msg + hb  (hb = h@Bw + dvec). grid (32, B), block 256.
// -------------------------------------------------------------------------
__global__ void big_gemm_kernel(const _Float16* __restrict__ Wh,
                                const _Float16* __restrict__ mpack,
                                const float* __restrict__ hb32,
                                _Float16* __restrict__ h16out,
                                float* __restrict__ out, int last) {
  extern __shared__ _Float16 smem[];  // 2 x MROWS x KCHUNK f16 (32 KB)
  const int mblock = blockIdx.x;      // 0..31
  const int b = blockIdx.y;
  const int wave = threadIdx.x >> 5;  // n-tile
  const int lane = threadIdx.x & 31;
  const int nIn = lane & 15;
  const int khalf = (lane < 16) ? 0 : 8;
  const _Float16* wbase = Wh + ((size_t)b * LSEQ + mblock * MROWS) * LSEQ;

  v8f acc[4] = {{}, {}, {}, {}};

  if (wave == 0) {
    tdm_load_tile(wbase, 0);
    __builtin_amdgcn_s_wait_tensorcnt(0);
  }
  __syncthreads();

  for (int c = 0; c < NCHUNKS; ++c) {
    if (wave == 0 && (c + 1) < NCHUNKS)
      tdm_load_tile(wbase + (size_t)(c + 1) * KCHUNK,
                    (unsigned)(((c + 1) & 1) * (MROWS * KCHUNK * 2)));
    const _Float16* tile = smem + (c & 1) * (MROWS * KCHUNK);
#pragma unroll
    for (int ks = 0; ks < 4; ++ks) {
      const int kt = c * 4 + ks;
      v16h bf = *(const v16h*)(mpack +
          (size_t)(((b * 64 + kt) * 8 + wave) * 32 + lane) * 16);
      const _Float16* ar = tile + (lane & 15) * KCHUNK + ks * 32 + khalf;
#pragma unroll
      for (int mt = 0; mt < 4; ++mt) {
        v8h lo = *(const v8h*)(ar + mt * 16 * KCHUNK);
        v8h hi = *(const v8h*)(ar + mt * 16 * KCHUNK + 16);
        v16h a = __builtin_shufflevector(lo, hi, 0, 1, 2, 3, 4, 5, 6, 7, 8, 9,
                                         10, 11, 12, 13, 14, 15);
        acc[mt] = __builtin_amdgcn_wmma_f32_16x16x32_f16(
            false, a, false, bf, (short)0, acc[mt], false, false);
      }
    }
    if (wave == 0 && (c + 1) < NCHUNKS) __builtin_amdgcn_s_wait_tensorcnt(0);
    __syncthreads();
  }

#pragma unroll
  for (int mt = 0; mt < 4; ++mt) {
#pragma unroll
    for (int v = 0; v < 8; ++v) {
      const int ii = mblock * MROWS + mt * 16 + v + 8 * (lane >> 4);
      const size_t idx = ((size_t)(b * LSEQ + ii)) * DD + wave * 16 + nIn;
      const float val = acc[mt][v] + hb32[idx];
      h16out[idx] = (_Float16)val;
      if (last) out[idx] = val;
    }
  }
}

// -------------------------------------------------------------------------
extern "C" void kernel_launch(void* const* d_in, const int* in_sizes, int n_in,
                              void* d_out, int out_size, void* d_ws,
                              size_t ws_size, hipStream_t stream) {
  (void)in_sizes; (void)n_in; (void)out_size; (void)ws_size;
  const float* coords   = (const float*)d_in[0];
  const int*   Z        = (const int*)d_in[1];
  const float* atom_emb = (const float*)d_in[2];
  const float* gamma    = (const float*)d_in[3];
  const float* centers  = (const float*)d_in[4];
  const float* rbf_w    = (const float*)d_in[5];
  const float* rbf_b    = (const float*)d_in[6];
  const float* out_w    = (const float*)d_in[7];
  const float* out_b    = (const float*)d_in[8];
  const float* self_w   = (const float*)d_in[9];
  const float* self_b   = (const float*)d_in[10];
  const float* msg_w    = (const float*)d_in[11];
  const float* msg_b    = (const float*)d_in[12];
  const float* upd_w    = (const float*)d_in[13];
  const float* upd_b    = (const float*)d_in[14];
  float* out = (float*)d_out;

  char* base = (char*)d_ws;
  size_t off = 0;
  auto alloc = [&](size_t bytes) -> void* {
    void* r = base + off;
    off = (off + bytes + 255) & ~(size_t)255;
    return r;
  };
  _Float16* Wh     = (_Float16*)alloc((size_t)NBATCH * LSEQ * LSEQ * 2); // 16.8 MB
  _Float16* h16    = (_Float16*)alloc((size_t)NBATCH * LSEQ * DD * 2);
  _Float16* mpack  = (_Float16*)alloc((size_t)NBATCH * LSEQ * DD * 2);
  float*    hb32   = (float*)alloc((size_t)NBATCH * LSEQ * DD * 4);
  float*    rbfloc = (float*)alloc((size_t)NBATCH * LSEQ * NRBF * 4);
  _Float16* Apack  = (_Float16*)alloc((size_t)4 * 8 * 32 * 16 * 2);
  _Float16* Bpack  = (_Float16*)alloc((size_t)4 * 8 * 32 * 16 * 2);
  float*    cvec   = (float*)alloc(DD * 4);
  float*    dvec   = (float*)alloc(DD * 4);

  dist_rbf_kernel<<<dim3(LSEQ, NBATCH), 256, 0, stream>>>(coords, gamma,
                                                          centers, Wh, rbfloc);
  encoder_kernel<<<dim3(LSEQ, NBATCH), 128, 0, stream>>>(
      Z, atom_emb, rbf_w, rbf_b, out_w, out_b, rbfloc, h16);

  for (int k = 0; k < KLAYERS; ++k) {
    fold_pack_kernel<<<dim3(32, 2), 256, 0, stream>>>(
        k, self_w, self_b, msg_w, msg_b, upd_w, upd_b, Apack, Bpack, cvec,
        dvec);
    med_gemm_kernel<<<dim3(256, 2), 256, 0, stream>>>(h16, Apack, Bpack, cvec,
                                                      dvec, mpack, hb32);
    big_gemm_kernel<<<dim3(32, NBATCH), 256,
                      2 * MROWS * KCHUNK * sizeof(_Float16), stream>>>(
        Wh, mpack, hb32, h16, out, (k == KLAYERS - 1) ? 1 : 0);
  }
}
